// _AttentionPool_12773232738461
// MI455X (gfx1250) — matrix operand order repaired and checked by reading, not compiled, not run on hardware
//
#include <hip/hip_runtime.h>
#include <hip/hip_bf16.h>

typedef __attribute__((ext_vector_type(16))) _Float16 v16h;
typedef __attribute__((ext_vector_type(8)))  _Float16 v8h;
typedef __attribute__((ext_vector_type(4)))  _Float16 v4h;
typedef __attribute__((ext_vector_type(8)))  float    v8f;

#define D_MODEL 1024
#define N_HEADS 16
#define D_KEY   64
#define BB      2
#define SS      2048

// ---------------------------------------------------------------------------
// Kernel 0: one-time f32 -> f16 conversion (states, Wv, Wo) so the GEMM hot
// loops stream half the bytes and carry zero cvt VALU work.
// ---------------------------------------------------------------------------
__global__ __launch_bounds__(256) void k_cvt16(const float* __restrict__ src,
                                               _Float16* __restrict__ dst,
                                               int n4) {
  const int i = blockIdx.x * 256 + threadIdx.x;
  if (i >= n4) return;
  const float4 f = ((const float4*)src)[i];
  v4h h;
  h[0] = (_Float16)f.x; h[1] = (_Float16)f.y;
  h[2] = (_Float16)f.z; h[3] = (_Float16)f.w;
  ((v4h*)dst)[i] = h;
}

// ---------------------------------------------------------------------------
// Kernel 1: v = states @ Wv^T + bv, stored transposed as f16 v_t[B][H][dk][S]
// One wave computes a 16(row) x 64(col) tile via 4 WMMA accumulators.
// A/B operands are pre-converted f16 (st16 / Wv16).
// ---------------------------------------------------------------------------
__global__ __launch_bounds__(128) void k_vproj(const _Float16* __restrict__ st16,
                                               const _Float16* __restrict__ Wv16,
                                               const float* __restrict__ bv,
                                               _Float16* __restrict__ v_t) {
  const int lane  = threadIdx.x & 31;
  const int wave  = threadIdx.x >> 5;              // 0..3
  const int r0    = blockIdx.y * 16;               // global row tile base
  const int n0    = (blockIdx.x * 4 + wave) * 64;  // output col group base
  const int lrow  = lane & 15;
  const int khalf = lane >> 4;

  v8f acc[4];
  #pragma unroll
  for (int t = 0; t < 4; ++t) {
    float bb = bv[n0 + t * 16 + lrow];
    #pragma unroll
    for (int m = 0; m < 8; ++m) acc[t][m] = bb;
  }

  const _Float16* arow = st16 + (size_t)(r0 + lrow) * D_MODEL;

  for (int kk = 0; kk < D_MODEL; kk += 32) {
    // A tile: lane = row, K chunks {0..15} (lanes 0-15) / {16..31}
    v8h a0 = *(const v8h*)(arow + kk + khalf * 16);
    v8h a1 = *(const v8h*)(arow + kk + khalf * 16 + 8);
    v16h a = __builtin_shufflevector(a0, a1, 0, 1, 2, 3, 4, 5, 6, 7,
                                     8, 9, 10, 11, 12, 13, 14, 15);
    #pragma unroll
    for (int t = 0; t < 4; ++t) {
      // B tile: lane = output col n, holds Wv[n][kk + khalf*16 .. +15]
      v16h bm = *(const v16h*)(Wv16 + (size_t)(n0 + t * 16 + lrow) * D_MODEL +
                               kk + khalf * 16);
      acc[t] = __builtin_amdgcn_wmma_f32_16x16x32_f16(false, a, false, bm,
                                                      (short)0, acc[t], false, false);
    }
  }

  // Store transposed: col n = h*64+d -> v_t[(b*1024 + n)*S + q], q contiguous
  const int b     = r0 >> 11;
  const int qbase = (r0 & (SS - 1)) + khalf * 8;
  #pragma unroll
  for (int t = 0; t < 4; ++t) {
    v8h pk;
    #pragma unroll
    for (int m = 0; m < 8; ++m) pk[m] = (_Float16)acc[t][m];
    *(v8h*)(v_t + ((size_t)(b * 1024 + n0 + t * 16 + lrow)) * SS + qbase) = pk;
  }
}

// ---------------------------------------------------------------------------
// Kernel 2: a[b,h,k] = states[b,k,:]·Wa[h,:]  (stored [B,H,S])
//           g[b,q,h] = states[b,q,:]·Wg[h,:] + bg[h]  (stored [B,S,H])
// Row staged in LDS via async-to-LDS b128 (ASYNCcnt), shuffle-reduced.
// ---------------------------------------------------------------------------
__global__ __launch_bounds__(256) void k_ag(const float* __restrict__ states,
                                            const float* __restrict__ Wa,
                                            const float* __restrict__ Wg,
                                            const float* __restrict__ bg,
                                            float* __restrict__ a_ws,
                                            float* __restrict__ g_ws) {
  __shared__ float srow[D_MODEL];
  const int r   = blockIdx.x;      // 0..B*S-1
  const int tid = threadIdx.x;
  const float* src = states + (size_t)r * D_MODEL;

  // each thread async-copies one 16B segment into LDS (one b128 per wave)
  {
    unsigned laddr = (unsigned)(uintptr_t)(&srow[0]) + tid * 16;
    const float* ga = src + tid * 4;
    asm volatile("global_load_async_to_lds_b128 %0, %1, off"
                 :: "v"(laddr), "v"(ga) : "memory");
    asm volatile("s_wait_asynccnt 0x0" ::: "memory");
  }
  __syncthreads();

  const int out = tid >> 3;        // 0..31  (0..15 -> a, 16..31 -> g)
  const int sub = tid & 7;
  const float* w = (out < 16) ? (Wa + (size_t)out * D_MODEL)
                              : (Wg + (size_t)(out - 16) * D_MODEL);
  float part = 0.f;
  const int base = sub * 128;
  #pragma unroll 8
  for (int j = 0; j < 128; ++j) part += srow[base + j] * w[base + j];

  part += __shfl_down(part, 4);
  part += __shfl_down(part, 2);
  part += __shfl_down(part, 1);
  if (sub == 0) {
    const int b = r >> 11, k = r & (SS - 1);
    if (out < 16) a_ws[((size_t)(b * N_HEADS + out)) * SS + k] = part;
    else          g_ws[(size_t)r * N_HEADS + (out - 16)] = part + bg[out - 16];
  }
}

// ---------------------------------------------------------------------------
// Kernel 3: flash-style attention.  One wave per (b, h, 16-row q tile).
// The mask tile (16x32 f32 = 2KB/chunk) is double-buffered in LDS via
// global_load_async_to_lds_b128: issue chunk kk+32 while consuming chunk kk,
// steady-state s_wait_asynccnt 0x4.  P@V via 4 WMMAs per chunk.
// ---------------------------------------------------------------------------
__global__ __launch_bounds__(32) void k_attn(const float* __restrict__ mask,
                                             const float* __restrict__ a_ws,
                                             const float* __restrict__ g_ws,
                                             const float* __restrict__ head_scale,
                                             const _Float16* __restrict__ v_t,
                                             _Float16* __restrict__ o_ws) {
  __shared__ float mt[2][16][32];          // 2 x 2KB mask tile buffers
  const int lane  = threadIdx.x;
  const int qt    = blockIdx.x;    // 0..127
  const int h     = blockIdx.y;    // 0..15
  const int b     = blockIdx.z;    // 0..1
  const int lrow  = lane & 15;
  const int khalf = lane >> 4;
  const int q0    = qt * 16;

  const float hs = head_scale[h];
  const float* mbase = mask + ((size_t)b * SS + q0) * SS;        // tile row base
  const float* arow  = a_ws + (size_t)(b * N_HEADS + h) * SS;
  const _Float16* vb = v_t + (size_t)(b * 1024 + h * D_KEY) * SS;
  const unsigned ldsb = (unsigned)(uintptr_t)(&mt[0][0][0]);

  // issue async copy of a 16x32 f32 mask chunk into buffer `bf`
  auto issue = [&](int kk, int bf) {
    #pragma unroll
    for (int j = 0; j < 4; ++j) {
      const int seg = lane + j * 32;               // 128 x 16B segments
      const unsigned laddr = ldsb + (unsigned)(bf * 2048 + seg * 16);
      const float* ga = mbase + (size_t)(seg >> 3) * SS + kk + (seg & 7) * 4;
      asm volatile("global_load_async_to_lds_b128 %0, %1, off"
                   :: "v"(laddr), "v"(ga) : "memory");
    }
  };

  v8f acc[4];
  #pragma unroll
  for (int t = 0; t < 4; ++t)
    #pragma unroll
    for (int m = 0; m < 8; ++m) acc[t][m] = 0.f;

  float m_run = -1e30f, l_run = 0.f;
  int bf = 0;
  issue(0, 0);

  for (int kk = 0; kk < SS; kk += 32) {
    if (kk + 32 < SS) {
      issue(kk + 32, bf ^ 1);
      asm volatile("s_wait_asynccnt 0x4" ::: "memory");  // current buffer ready
    } else {
      asm volatile("s_wait_asynccnt 0x0" ::: "memory");
    }

    const float* tp  = &mt[bf][lrow][khalf * 8];
    const float* a0p = arow + kk + khalf * 8;

    float lg[16];
    #pragma unroll
    for (int i = 0; i < 8; ++i) {
      lg[i]     = a0p[i]      + hs * tp[i];        // K = khalf*8 + i
      lg[8 + i] = a0p[16 + i] + hs * tp[16 + i];   // K = 16 + khalf*8 + i
    }
    float cm = lg[0];
    #pragma unroll
    for (int i = 1; i < 16; ++i) cm = fmaxf(cm, lg[i]);
    cm = fmaxf(cm, __shfl_xor(cm, 16));            // combine row halves
    const float nm    = fmaxf(m_run, cm);
    const float alpha = __expf(m_run - nm);

    v16h p;
    float rs = 0.f;
    #pragma unroll
    for (int i = 0; i < 16; ++i) {
      float e = __expf(lg[i] - nm);
      rs += e;
      p[i] = (_Float16)e;
    }
    rs += __shfl_xor(rs, 16);
    l_run = l_run * alpha + rs;
    m_run = nm;

    // broadcast per-row alpha (lane = row) into C layout (VGPR = row)
    float al[8];
    #pragma unroll
    for (int m = 0; m < 8; ++m) al[m] = __shfl(alpha, m + khalf * 8);
    #pragma unroll
    for (int t = 0; t < 4; ++t)
      #pragma unroll
      for (int m = 0; m < 8; ++m) acc[t][m] *= al[m];

    #pragma unroll
    for (int t = 0; t < 4; ++t) {
      // V tile: lane = col d, k chunks {khalf*8..+7, 16+khalf*8..+7} (v_t transposed layout)
      const _Float16* vp = vb + (size_t)(t * 16 + lrow) * SS + kk + khalf * 8;
      v8h b0 = *(const v8h*)vp;
      v8h b1 = *(const v8h*)(vp + 16);
      v16h bm = __builtin_shufflevector(b0, b1, 0, 1, 2, 3, 4, 5, 6, 7,
                                        8, 9, 10, 11, 12, 13, 14, 15);
      acc[t] = __builtin_amdgcn_wmma_f32_16x16x32_f16(false, p, false, bm,
                                                      (short)0, acc[t], false, false);
    }
    bf ^= 1;
  }

  // epilogue: scale each row by sigmoid(g)/l_run
  const float linv = 1.0f / l_run;
  const float* gp = g_ws + ((size_t)(b * SS + q0)) * N_HEADS + h;
  float sc[8];
  #pragma unroll
  for (int m = 0; m < 8; ++m) {
    float li = __shfl(linv, m + khalf * 8);
    float gv = gp[(size_t)(khalf * 8 + m) * N_HEADS];
    sc[m] = li * (1.f / (1.f + __expf(-gv)));
  }
  #pragma unroll
  for (int t = 0; t < 4; ++t) {
    #pragma unroll
    for (int m = 0; m < 8; ++m) {
      const int q = q0 + khalf * 8 + m;
      o_ws[((size_t)(b * SS + q)) * D_MODEL + h * D_KEY + t * 16 + lrow] =
          (_Float16)(acc[t][m] * sc[m]);
    }
  }
}

// ---------------------------------------------------------------------------
// Kernel 4: out = o @ Wo^T + bo  (f16 A and B operands, f32 output)
// ---------------------------------------------------------------------------
__global__ __launch_bounds__(128) void k_oproj(const _Float16* __restrict__ o_ws,
                                               const _Float16* __restrict__ Wo16,
                                               const float* __restrict__ bo,
                                               float* __restrict__ out) {
  const int lane  = threadIdx.x & 31;
  const int wave  = threadIdx.x >> 5;
  const int r0    = blockIdx.y * 16;
  const int n0    = (blockIdx.x * 4 + wave) * 64;
  const int lrow  = lane & 15;
  const int khalf = lane >> 4;

  v8f acc[4];
  #pragma unroll
  for (int t = 0; t < 4; ++t) {
    float bb = bo[n0 + t * 16 + lrow];
    #pragma unroll
    for (int m = 0; m < 8; ++m) acc[t][m] = bb;
  }

  const _Float16* arow = o_ws + (size_t)(r0 + lrow) * D_MODEL;
  for (int kk = 0; kk < D_MODEL; kk += 32) {
    v8h a0 = *(const v8h*)(arow + kk + khalf * 16);
    v8h a1 = *(const v8h*)(arow + kk + khalf * 16 + 8);
    v16h a = __builtin_shufflevector(a0, a1, 0, 1, 2, 3, 4, 5, 6, 7,
                                     8, 9, 10, 11, 12, 13, 14, 15);
    #pragma unroll
    for (int t = 0; t < 4; ++t) {
      v16h bm = *(const v16h*)(Wo16 + (size_t)(n0 + t * 16 + lrow) * D_MODEL +
                               kk + khalf * 16);
      acc[t] = __builtin_amdgcn_wmma_f32_16x16x32_f16(false, a, false, bm,
                                                      (short)0, acc[t], false, false);
    }
  }
  #pragma unroll
  for (int t = 0; t < 4; ++t) {
    #pragma unroll
    for (int m = 0; m < 8; ++m) {
      const int r = r0 + khalf * 8 + m;
      out[(size_t)r * D_MODEL + n0 + t * 16 + lrow] = acc[t][m];
    }
  }
}

// ---------------------------------------------------------------------------
// Launch
// ---------------------------------------------------------------------------
extern "C" void kernel_launch(void* const* d_in, const int* in_sizes, int n_in,
                              void* d_out, int out_size, void* d_ws, size_t ws_size,
                              hipStream_t stream) {
  const float* states = (const float*)d_in[0];
  const float* mask   = (const float*)d_in[1];
  const float* hscale = (const float*)d_in[2];
  const float* Wa     = (const float*)d_in[3];
  const float* Wv     = (const float*)d_in[4];
  const float* bv     = (const float*)d_in[5];
  const float* Wg     = (const float*)d_in[6];
  const float* bg     = (const float*)d_in[7];
  const float* Wo     = (const float*)d_in[8];
  const float* bo     = (const float*)d_in[9];
  float* out = (float*)d_out;
  (void)in_sizes; (void)n_in; (void)out_size; (void)ws_size;

  char* ws = (char*)d_ws;
  _Float16* v_t  = (_Float16*)ws;                               // 8 MiB: [B,H,dk,S] f16
  _Float16* o_ws = (_Float16*)(ws + (8u << 20));                // 8 MiB: [B*S,1024] f16
  float*    a_ws = (float*)(ws + (16u << 20));                  // 256 KiB: [B,H,S]
  float*    g_ws = (float*)(ws + (16u << 20) + (256u << 10));   // 256 KiB: [B,S,H]
  _Float16* st16 = (_Float16*)(ws + (17u << 20));               // 8 MiB: states f16
  _Float16* Wv16 = (_Float16*)(ws + (25u << 20));               // 2 MiB
  _Float16* Wo16 = (_Float16*)(ws + (27u << 20));               // 2 MiB

  // one-time f32 -> f16 conversions (each thread handles 4 elements)
  k_cvt16<<<dim3((BB * SS * D_MODEL / 4 + 255) / 256), 256, 0, stream>>>(states, st16, BB * SS * D_MODEL / 4);
  k_cvt16<<<dim3((D_MODEL * D_MODEL / 4 + 255) / 256), 256, 0, stream>>>(Wv, Wv16, D_MODEL * D_MODEL / 4);
  k_cvt16<<<dim3((D_MODEL * D_MODEL / 4 + 255) / 256), 256, 0, stream>>>(Wo, Wo16, D_MODEL * D_MODEL / 4);

  k_vproj<<<dim3(4, 256), 128, 0, stream>>>(st16, Wv16, bv, v_t);
  k_ag   <<<dim3(BB * SS), 256, 0, stream>>>(states, Wa, Wg, bg, a_ws, g_ws);
  k_attn <<<dim3(SS / 16, N_HEADS, BB), 32, 0, stream>>>(mask, a_ws, g_ws, hscale, v_t, o_ws);
  k_oproj<<<dim3(4, 256), 128, 0, stream>>>(o_ws, Wo16, bo, out);
}
